// RelationModule_14886356648415
// MI455X (gfx1250) — matrix operand order repaired and checked by reading, not compiled, not run on hardware
//
#include <hip/hip_runtime.h>
#include <math.h>

typedef __attribute__((ext_vector_type(16))) _Float16 v16h;
typedef __attribute__((ext_vector_type(8)))  float    v8f;

constexpr int kB  = 8;
constexpr int kN  = 256;
constexpr int kP  = 40000;
constexpr int kNS = 1024;
constexpr int kHID = 128;

#define WMMA_F16(a, b, c) \
  __builtin_amdgcn_wmma_f32_16x16x32_f16(false, (a), false, (b), (short)0, (c), false, false)

// ---------------------------------------------------------------------------
// A fragment: 16x32 (MxK). Per lane: two contiguous 8-float runs at
// Arow[kbase + hi*8] and Arow[kbase + 16 + hi*8].
template <int K>
__device__ __forceinline__ v16h load_a_frag(const float* __restrict__ Arow,
                                            int kbase, int hi) {
  v16h a;
  if constexpr ((K & 31) == 0) {
    const float* p = Arow + kbase + (hi << 3);
    float4 a0 = *(const float4*)(p);
    float4 a1 = *(const float4*)(p + 4);
    float4 a2 = *(const float4*)(p + 16);
    float4 a3 = *(const float4*)(p + 20);
    a[0] = (_Float16)a0.x; a[1] = (_Float16)a0.y; a[2] = (_Float16)a0.z; a[3] = (_Float16)a0.w;
    a[4] = (_Float16)a1.x; a[5] = (_Float16)a1.y; a[6] = (_Float16)a1.z; a[7] = (_Float16)a1.w;
    a[8]  = (_Float16)a2.x; a[9]  = (_Float16)a2.y; a[10] = (_Float16)a2.z; a[11] = (_Float16)a2.w;
    a[12] = (_Float16)a3.x; a[13] = (_Float16)a3.y; a[14] = (_Float16)a3.z; a[15] = (_Float16)a3.w;
  } else {
    // Ragged K: clamp index (always in-bounds load) + select value to zero.
#pragma unroll
    for (int e = 0; e < 16; ++e) {
      int k = kbase + ((e >= 8) ? 16 : 0) + (hi << 3) + (e & 7);
      int kc = (k < K) ? k : 0;
      float v = Arow[kc];
      a[e] = (_Float16)((k < K) ? v : 0.0f);
    }
  }
  return a;
}

// B fragment: 32x16 (KxN). Lane holds column n; two K-runs of 8 at
// kbase + hi*8 and kbase + 16 + hi*8. Element [k][n] at W[k*ks + n*ns].
template <int K>
__device__ __forceinline__ v16h load_b_frag(const float* __restrict__ W,
                                            long ks, long ns, int n,
                                            int kbase, int hi) {
  v16h b;
  if constexpr ((K & 31) == 0) {
    const float* p = W + (long)(kbase + (hi << 3)) * ks + (long)n * ns;
#pragma unroll
    for (int e = 0; e < 16; ++e) b[e] = (_Float16)p[(long)(e + ((e >= 8) ? 8 : 0)) * ks];
  } else {
#pragma unroll
    for (int e = 0; e < 16; ++e) {
      int k = kbase + ((e >= 8) ? 16 : 0) + (hi << 3) + (e & 7);
      int kc = (k < K) ? k : 0;
      float v = W[(long)kc * ks + (long)n * ns];
      b[e] = (_Float16)((k < K) ? v : 0.0f);
    }
  }
  return b;
}

// ---------------------------------------------------------------------------
// Generic WMMA GEMM: one wave per 16x16 output tile.
// MODE 0: out = acc + bias
// MODE 1: out = prelu(bn(acc + bias))        (conv1 epilogue)
// MODE 2: out = Cprev + alpha*(acc + bias)   (residual accumulate)
template <int K, int MODE>
__global__ void gemm_wmma_kernel(const float* __restrict__ A, int lda,
                                 const float* __restrict__ W,
                                 int M, int Ncols,
                                 const float* __restrict__ bias,
                                 const float* __restrict__ Cprev, float alpha,
                                 float* __restrict__ Cout,
                                 const float* __restrict__ bn_g,
                                 const float* __restrict__ bn_b,
                                 const float* __restrict__ bn_m,
                                 const float* __restrict__ bn_v,
                                 const float* __restrict__ pa) {
  int wave = threadIdx.x >> 5, lane = threadIdx.x & 31;
  int hi = lane >> 4, ln16 = lane & 15;
  int ntn = Ncols >> 4;
  int tile = blockIdx.x * (blockDim.x >> 5) + wave;
  int total = (M >> 4) * ntn;
  if (tile >= total) return;
  int tm = tile / ntn, tn = tile % ntn;
  int m0 = tm << 4, n0 = tn << 4;
  int col = n0 + ln16;

  if constexpr (MODE == 2) {
    // Overlap the residual-tile read with the K-loop (global_prefetch_b8).
    __builtin_prefetch(&Cprev[(size_t)(m0 + (hi << 3)) * Ncols + col], 0, 3);
    __builtin_prefetch(&Cprev[(size_t)(m0 + (hi << 3) + 7) * Ncols + col], 0, 3);
  }

  v8f acc = {};
  const float* Arow = A + (size_t)(m0 + ln16) * lda;
#pragma unroll
  for (int k0 = 0; k0 < K; k0 += 32) {
    v16h af = load_a_frag<K>(Arow, k0, hi);
    v16h bf = load_b_frag<K>(W, (long)Ncols, 1, n0 + ln16, k0, hi);
    acc = WMMA_F16(af, bf, acc);
  }

  float bv = bias[col];
#pragma unroll
  for (int i = 0; i < 8; ++i) {
    int row = m0 + (hi << 3) + i;
    float v = acc[i] + bv;
    if constexpr (MODE == 1) {
      v = (v - bn_m[col]) * rsqrtf(bn_v[col] + 1e-5f) * bn_g[col] + bn_b[col];
      v = (v >= 0.0f) ? v : pa[col] * v;
    } else if constexpr (MODE == 2) {
      v = Cprev[(size_t)row * Ncols + col] + alpha * v;
    }
    Cout[(size_t)row * Ncols + col] = v;
  }
}

// ---------------------------------------------------------------------------
// Per-box precompute: centers, sizes, bbox_in(27), obj_rows gather (128).
__global__ void precompute_kernel(const float* __restrict__ corners,
                                  const float* __restrict__ pc,
                                  const int* __restrict__ seed_inds,
                                  const int* __restrict__ avi,
                                  float* __restrict__ centers,
                                  float* __restrict__ sizes,
                                  float* __restrict__ bbox_in,
                                  float* __restrict__ obj_rows) {
  int bn = blockIdx.x;              // 0..B*N-1
  int b = bn >> 8;
  int tid = threadIdx.x;            // 128 threads
  __shared__ float c3[3];
  const float* C = corners + (size_t)bn * 24;
  if (tid == 0) {
    float mn[3], mx[3];
#pragma unroll
    for (int d = 0; d < 3; ++d) { mn[d] = C[d]; mx[d] = C[d]; }
#pragma unroll
    for (int k = 1; k < 8; ++k)
#pragma unroll
      for (int d = 0; d < 3; ++d) {
        float v = C[k * 3 + d];
        mn[d] = fminf(mn[d], v);
        mx[d] = fmaxf(mx[d], v);
      }
#pragma unroll
    for (int d = 0; d < 3; ++d) {
      float c = (mn[d] + mx[d]) * 0.5f;
      centers[bn * 3 + d] = c;
      sizes[bn * 3 + d] = mx[d] - mn[d];
      c3[d] = c;
    }
  }
  __syncthreads();
  if (tid < 27) {
    float v = (tid < 3) ? c3[tid] : (C[tid - 3] - c3[(tid - 3) % 3]);
    bbox_in[(size_t)bn * 27 + tid] = v;
  }
  // obj_feat = swapaxes(pc[...,6:134],1,2) -> (B,128,P); gathered rows of its
  // flat (-1,128) reshape.
  int r = seed_inds[b * kNS + avi[bn]] + b * 128;
  long f = (long)r * 128 + tid;
  int bb = (int)(f / (128L * kP));
  long rem = f - (long)bb * (128L * kP);
  int c = (int)(rem / kP);
  int p = (int)(rem - (long)c * kP);
  obj_rows[(size_t)bn * 128 + tid] = pc[((size_t)bb * kP + p) * 135 + 6 + c];
}

// ---------------------------------------------------------------------------
// knn mask: one thread per (b,i) row; 16 nearest (squared dist, first-index
// tie-break like top_k) get 0.0, rest -10000.
__global__ void knn_mask_kernel(const float* __restrict__ centers,
                                float* __restrict__ mask) {
  int t = blockIdx.x * blockDim.x + threadIdx.x;
  if (t >= kB * kN) return;
  int b = t >> 8, i = t & 255;
  const float* cb = centers + (size_t)b * kN * 3;
  float cx = cb[i * 3], cy = cb[i * 3 + 1], cz = cb[i * 3 + 2];
  float d[256];
  for (int j = 0; j < kN; ++j) {
    float dx = cb[j * 3] - cx, dy = cb[j * 3 + 1] - cy, dz = cb[j * 3 + 2] - cz;
    d[j] = dx * dx + dy * dy + dz * dz;
  }
  float* mrow = mask + (size_t)t * kN;
  for (int j = 0; j < kN; ++j) mrow[j] = -10000.0f;
  for (int s = 0; s < 16; ++s) {
    float best = 3.0e38f;
    int bj = 0;
    for (int j = 0; j < kN; ++j)
      if (d[j] < best) { best = d[j]; bj = j; }
    mrow[bj] = 0.0f;
    d[bj] = 3.0e38f;
  }
}

// ---------------------------------------------------------------------------
// Fused geometry MLP: per wave one (b,i, j0..j0+15) strip.
// geom(12) -> fc1(12x64) via WMMA -> +bias,relu -> LN(64) -> fc2(64x4)
// -> att_bias[b,h,i,j] = bias + knn_mask.
__global__ void geom_mlp_kernel(const float* __restrict__ centers,
                                const float* __restrict__ sizes,
                                const float* __restrict__ fc1_w,
                                const float* __restrict__ fc1_b,
                                const float* __restrict__ ln_g,
                                const float* __restrict__ ln_b,
                                const float* __restrict__ fc2_w,
                                const float* __restrict__ fc2_b,
                                const float* __restrict__ knn_mask,
                                float* __restrict__ att_bias) {
  __shared__ float hbuf[8][16][64];
  int wave = threadIdx.x >> 5, lane = threadIdx.x & 31;
  int hi = lane >> 4, ln16 = lane & 15;
  int tile = blockIdx.x * 8 + wave;   // exactly B*N*N/16 tiles
  int j0 = (tile & 15) << 4;
  int i = (tile >> 4) & 255;
  int b = tile >> 12;
  int j = j0 + ln16;

  const float* cb = centers + (size_t)b * kN * 3;
  const float* sb = sizes + (size_t)b * kN * 3;
  float rx = cb[j * 3 + 0] - cb[i * 3 + 0];
  float ry = cb[j * 3 + 1] - cb[i * 3 + 1];
  float rz = cb[j * 3 + 2] - cb[i * 3 + 2];
  float dist = sqrtf(rx * rx + ry * ry + rz * rz);
  float horiz = sqrtf(rx * rx + ry * ry);
  float ce = rz / (dist + 1e-6f);
  float six = sb[i * 3], siy = sb[i * 3 + 1], siz = sb[i * 3 + 2];
  float sjx = sb[j * 3], sjy = sb[j * 3 + 1], sjz = sb[j * 3 + 2];
  float g12[12] = {rx, ry, rz, dist, horiz, ce,
                   sjx - six, sjy - siy, sjz - siz,
                   sjx / (six + 1e-6f), sjy / (siy + 1e-6f), sjz / (siz + 1e-6f)};

  // A fragment (K=12 padded to 32): hi=0 -> g12[e] for e<8; hi=1 -> g12[8+e] for e<4.
  v16h af;
#pragma unroll
  for (int e = 0; e < 16; ++e) {
    float v0 = (e < 8) ? g12[e] : 0.0f;
    float v1 = (e < 4) ? g12[8 + e] : 0.0f;
    af[e] = (_Float16)(hi ? v1 : v0);
  }

#pragma unroll
  for (int nt = 0; nt < 4; ++nt) {
    int n = nt * 16 + ln16;
    v16h bf;
#pragma unroll
    for (int e = 0; e < 16; ++e) {
      int k = ((e >= 8) ? 16 : 0) + (hi << 3) + (e & 7);
      int kc = (k < 12) ? k : 0;            // clamped in-bounds load
      float v = fc1_w[kc * 64 + n];
      bf[e] = (_Float16)((k < 12) ? v : 0.0f);
    }
    v8f acc = {};
    acc = WMMA_F16(af, bf, acc);
#pragma unroll
    for (int r = 0; r < 8; ++r) {
      int row = (hi << 3) + r;
      float v = acc[r] + fc1_b[n];
      hbuf[wave][row][n] = (v > 0.0f) ? v : 0.0f;   // relu
    }
  }
  __syncthreads();

  if (lane < 16) {
    const float* hr = hbuf[wave][lane];
    float mean = 0.0f;
    for (int n = 0; n < 64; ++n) mean += hr[n];
    mean *= (1.0f / 64.0f);
    float var = 0.0f;
    for (int n = 0; n < 64; ++n) { float dd = hr[n] - mean; var += dd * dd; }
    var *= (1.0f / 64.0f);
    float rinv = rsqrtf(var + 1e-5f);
    float o0 = fc2_b[0], o1 = fc2_b[1], o2 = fc2_b[2], o3 = fc2_b[3];
    for (int n = 0; n < 64; ++n) {
      float tn = (hr[n] - mean) * rinv * ln_g[n] + ln_b[n];
      o0 += tn * fc2_w[n * 4 + 0];
      o1 += tn * fc2_w[n * 4 + 1];
      o2 += tn * fc2_w[n * 4 + 2];
      o3 += tn * fc2_w[n * 4 + 3];
    }
    int jj = j0 + lane;
    float msk = knn_mask[((size_t)b * kN + i) * kN + jj];
    att_bias[(((size_t)b * 4 + 0) * kN + i) * kN + jj] = o0 + msk;
    att_bias[(((size_t)b * 4 + 1) * kN + i) * kN + jj] = o1 + msk;
    att_bias[(((size_t)b * 4 + 2) * kN + i) * kN + jj] = o2 + msk;
    att_bias[(((size_t)b * 4 + 3) * kN + i) * kN + jj] = o3 + msk;
  }
}

// ---------------------------------------------------------------------------
// att[b,h,q,k] = (Q.K^T)*scale + att_bias ; one wave per 16x16 tile, K=32.
__global__ void att_qk_kernel(const float* __restrict__ q,
                              const float* __restrict__ k,
                              const float* __restrict__ att_bias,
                              float* __restrict__ att, float scale) {
  int wave = threadIdx.x >> 5, lane = threadIdx.x & 31;
  int hi = lane >> 4, ln16 = lane & 15;
  int tile = blockIdx.x * 8 + wave;   // 8192 tiles
  int bh = tile >> 8;
  int tm = (tile >> 4) & 15, tn = tile & 15;
  int b = bh >> 2, h = bh & 3;
  const float* qb = q + (size_t)b * kN * kHID + h * 32;
  const float* kb = k + (size_t)b * kN * kHID + h * 32;
  int m0 = tm << 4, n0 = tn << 4;
  size_t idx0 = ((size_t)bh * kN + m0 + (hi << 3)) * kN + n0 + ln16;
  __builtin_prefetch(&att_bias[idx0], 0, 3);
  v16h af = load_a_frag<32>(qb + (size_t)(m0 + ln16) * kHID, 0, hi);
  v16h bf = load_b_frag<32>(kb, 1, kHID, n0 + ln16, 0, hi);  // B[d][kk]=k[kk*128+d]
  v8f acc = {};
  acc = WMMA_F16(af, bf, acc);
#pragma unroll
  for (int i = 0; i < 8; ++i) {
    size_t idx = idx0 + (size_t)i * kN;
    att[idx] = acc[i] * scale + att_bias[idx];
  }
}

// Row softmax over 256 entries; one block per (b,h,q) row.
__global__ void softmax_kernel(float* __restrict__ att) {
  __shared__ float red[256];
  int row = blockIdx.x, t = threadIdx.x;
  float* r = att + (size_t)row * kN;
  float x = r[t];
  red[t] = x;
  __syncthreads();
  for (int s = 128; s > 0; s >>= 1) {
    if (t < s) red[t] = fmaxf(red[t], red[t + s]);
    __syncthreads();
  }
  float mx = red[0];
  __syncthreads();
  float e = __expf(x - mx);
  red[t] = e;
  __syncthreads();
  for (int s = 128; s > 0; s >>= 1) {
    if (t < s) red[t] += red[t + s];
    __syncthreads();
  }
  r[t] = e / red[0];
}

// out[b,q,h*32+d] = sum_k att[b,h,q,k] * v[b,k,h,d] ; one wave per tile, K=256.
__global__ void att_v_kernel(const float* __restrict__ att,
                             const float* __restrict__ v,
                             float* __restrict__ out) {
  int wave = threadIdx.x >> 5, lane = threadIdx.x & 31;
  int hi = lane >> 4, ln16 = lane & 15;
  int tile = blockIdx.x * 8 + wave;   // 1024 tiles
  int bh = tile >> 5;
  int rest = tile & 31;
  int tm = rest >> 1, tn = rest & 1;
  int b = bh >> 2, h = bh & 3;
  const float* ab = att + (size_t)bh * kN * kN;
  const float* vb = v + (size_t)b * kN * kHID + h * 32;
  int m0 = tm << 4, n0 = tn << 4;
  v8f acc = {};
  const float* Arow = ab + (size_t)(m0 + ln16) * kN;
#pragma unroll
  for (int k0 = 0; k0 < 256; k0 += 32) {
    v16h af = load_a_frag<256>(Arow, k0, hi);
    v16h bf = load_b_frag<256>(vb, kHID, 1, n0 + ln16, k0, hi);  // B[kk][d]
    acc = WMMA_F16(af, bf, acc);
  }
#pragma unroll
  for (int i = 0; i < 8; ++i) {
    out[((size_t)b * kN + m0 + (hi << 3) + i) * kHID + h * 32 + n0 + ln16] = acc[i];
  }
}

// LayerNorm over width 128; one block (128 threads) per row.
__global__ void ln_kernel(const float* __restrict__ src,
                          const float* __restrict__ g,
                          const float* __restrict__ b,
                          float* __restrict__ dst) {
  __shared__ float red[128];
  int row = blockIdx.x, t = threadIdx.x;
  float x = src[(size_t)row * kHID + t];
  red[t] = x;
  __syncthreads();
  for (int s = 64; s > 0; s >>= 1) {
    if (t < s) red[t] += red[t + s];
    __syncthreads();
  }
  float mean = red[0] * (1.0f / 128.0f);
  __syncthreads();
  float d = x - mean;
  red[t] = d * d;
  __syncthreads();
  for (int s = 64; s > 0; s >>= 1) {
    if (t < s) red[t] += red[t + s];
    __syncthreads();
  }
  float var = red[0] * (1.0f / 128.0f);
  dst[(size_t)row * kHID + t] = d * rsqrtf(var + 1e-5f) * g[t] + b[t];
}

// ---------------------------------------------------------------------------
struct LayerP {
  const float *fc1_w, *fc1_b, *ln_g, *ln_b, *fc2_w, *fc2_b;
  const float *wq, *bq, *wk, *bk, *wv, *bv, *wo, *bo;
  const float *aln_g, *aln_b, *bbox_w, *bbox_b, *obj_w, *obj_b;
};

extern "C" void kernel_launch(void* const* d_in, const int* in_sizes, int n_in,
                              void* d_out, int out_size, void* d_ws, size_t ws_size,
                              hipStream_t stream) {
  (void)out_size; (void)ws_size; (void)n_in;
  auto F = [&](int i) { return (const float*)d_in[i]; };

  const float *pbf, *corners, *pc, *conv1_w, *conv1_b, *bn_g, *bn_b, *bn_m, *bn_v,
      *prelu_a, *conv2_w, *conv2_b;
  const int *seed, *avi;
  LayerP L[2];

  bool insertion = (in_sizes[0] == kB * kN * kHID);  // pred_bbox_feature first?
  if (insertion) {
    pbf = F(0); corners = F(1); pc = F(2);
    conv1_w = F(3); conv1_b = F(4);
    bn_g = F(5); bn_b = F(6); bn_m = F(7); bn_v = F(8);
    prelu_a = F(9); conv2_w = F(10); conv2_b = F(11);
    for (int l = 0; l < 2; ++l) {
      int o = 12 + l * 20;
      L[l].fc1_w = F(o + 0);  L[l].fc1_b = F(o + 1);
      L[l].ln_g = F(o + 2);   L[l].ln_b = F(o + 3);
      L[l].fc2_w = F(o + 4);  L[l].fc2_b = F(o + 5);
      L[l].wq = F(o + 6);  L[l].bq = F(o + 7);
      L[l].wk = F(o + 8);  L[l].bk = F(o + 9);
      L[l].wv = F(o + 10); L[l].bv = F(o + 11);
      L[l].wo = F(o + 12); L[l].bo = F(o + 13);
      L[l].aln_g = F(o + 14); L[l].aln_b = F(o + 15);
      L[l].bbox_w = F(o + 16); L[l].bbox_b = F(o + 17);
      L[l].obj_w = F(o + 18);  L[l].obj_b = F(o + 19);
    }
    seed = (const int*)d_in[52];
    avi = (const int*)d_in[53];
  } else {
    // jax pytree flatten: sorted dict keys at every level.
    avi = (const int*)d_in[0];
    bn_b = F(1); bn_g = F(2); bn_m = F(3); bn_v = F(4);
    conv1_b = F(5); conv1_w = F(6); conv2_b = F(7); conv2_w = F(8);
    for (int l = 0; l < 2; ++l) {
      int o = 9 + l * 20;
      L[l].aln_b = F(o + 0);  L[l].aln_g = F(o + 1);
      L[l].bbox_b = F(o + 2); L[l].bbox_w = F(o + 3);
      L[l].bk = F(o + 4); L[l].bo = F(o + 5); L[l].bq = F(o + 6); L[l].bv = F(o + 7);
      L[l].fc1_b = F(o + 8);  L[l].fc1_w = F(o + 9);
      L[l].fc2_b = F(o + 10); L[l].fc2_w = F(o + 11);
      L[l].ln_b = F(o + 12);  L[l].ln_g = F(o + 13);
      L[l].obj_b = F(o + 14); L[l].obj_w = F(o + 15);
      L[l].wk = F(o + 16); L[l].wo = F(o + 17); L[l].wq = F(o + 18); L[l].wv = F(o + 19);
    }
    prelu_a = F(49);
    pc = F(50); corners = F(51); pbf = F(52);
    seed = (const int*)d_in[53];
  }

  // ---- workspace carve-up (floats) ----
  float* w = (float*)d_ws;
  const size_t MN = (size_t)kB * kN;           // 2048
  float* x1       = w;                 w += MN * kHID;
  float* obj_rows = w;                 w += MN * kHID;
  float* bbox_in  = w;                 w += MN * 27;
  float* centers  = w;                 w += MN * 3;
  float* sizes    = w;                 w += MN * 3;
  float* knn_mask = w;                 w += MN * kN;
  float* att_bias = w;                 w += (size_t)kB * 4 * kN * kN;
  float* att      = w;                 w += (size_t)kB * 4 * kN * kN;
  float* qb       = w;                 w += MN * kHID;
  float* kb       = w;                 w += MN * kHID;
  float* vb       = w;                 w += MN * kHID;
  float* attn_out = w;                 w += MN * kHID;
  float* resid    = w;                 w += MN * kHID;
  float* features = (float*)d_out;     // final output lives here

  const float inv_sqrt_dk = 0.1767766952966369f;  // 1/sqrt(32)

  // 1) geometry precompute + gathers
  precompute_kernel<<<kB * kN, 128, 0, stream>>>(corners, pc, seed, avi,
                                                 centers, sizes, bbox_in, obj_rows);
  // 2) knn mask
  knn_mask_kernel<<<(kB * kN) / 256, 256, 0, stream>>>(centers, knn_mask);

  // 3) features = conv2( prelu(bn( pbf @ conv1 + b1 )) ) + b2
  int gemm_blocks = ((int)(MN / 16) * (kHID / 16) + 7) / 8;   // 128
  gemm_wmma_kernel<128, 1><<<gemm_blocks, 256, 0, stream>>>(
      pbf, kHID, conv1_w, (int)MN, kHID, conv1_b,
      nullptr, 0.0f, x1, bn_g, bn_b, bn_m, bn_v, prelu_a);
  gemm_wmma_kernel<128, 0><<<gemm_blocks, 256, 0, stream>>>(
      x1, kHID, conv2_w, (int)MN, kHID, conv2_b,
      nullptr, 0.0f, features, nullptr, nullptr, nullptr, nullptr, nullptr);

  for (int l = 0; l < 2; ++l) {
    const LayerP& lp = L[l];
    // 4) fused geom MLP -> attention bias (incl. knn mask)
    geom_mlp_kernel<<<(kB * kN * kN / 16) / 8, 256, 0, stream>>>(
        centers, sizes, lp.fc1_w, lp.fc1_b, lp.ln_g, lp.ln_b,
        lp.fc2_w, lp.fc2_b, knn_mask, att_bias);
    // 5) features += 0.1*(obj_rows @ obj_w + obj_b)
    gemm_wmma_kernel<128, 2><<<gemm_blocks, 256, 0, stream>>>(
        obj_rows, kHID, lp.obj_w, (int)MN, kHID, lp.obj_b,
        features, 0.1f, features, nullptr, nullptr, nullptr, nullptr, nullptr);
    // 6) features += bbox_in @ bbox_w + bbox_b   (K=27, zero-padded)
    gemm_wmma_kernel<27, 2><<<gemm_blocks, 256, 0, stream>>>(
        bbox_in, 27, lp.bbox_w, (int)MN, kHID, lp.bbox_b,
        features, 1.0f, features, nullptr, nullptr, nullptr, nullptr, nullptr);
    // 7) q,k,v
    gemm_wmma_kernel<128, 0><<<gemm_blocks, 256, 0, stream>>>(
        features, kHID, lp.wq, (int)MN, kHID, lp.bq,
        nullptr, 0.0f, qb, nullptr, nullptr, nullptr, nullptr, nullptr);
    gemm_wmma_kernel<128, 0><<<gemm_blocks, 256, 0, stream>>>(
        features, kHID, lp.wk, (int)MN, kHID, lp.bk,
        nullptr, 0.0f, kb, nullptr, nullptr, nullptr, nullptr, nullptr);
    gemm_wmma_kernel<128, 0><<<gemm_blocks, 256, 0, stream>>>(
        features, kHID, lp.wv, (int)MN, kHID, lp.bv,
        nullptr, 0.0f, vb, nullptr, nullptr, nullptr, nullptr, nullptr);
    // 8) attention
    att_qk_kernel<<<(kB * 4 * 16 * 16) / 8, 256, 0, stream>>>(qb, kb, att_bias, att,
                                                              inv_sqrt_dk);
    softmax_kernel<<<kB * 4 * kN, 256, 0, stream>>>(att);
    att_v_kernel<<<(kB * 4 * 32) / 8, 256, 0, stream>>>(att, vb, attn_out);
    // 9) resid = features + attn_out @ wo + bo ; features = LN(resid)
    gemm_wmma_kernel<128, 2><<<gemm_blocks, 256, 0, stream>>>(
        attn_out, kHID, lp.wo, (int)MN, kHID, lp.bo,
        features, 1.0f, resid, nullptr, nullptr, nullptr, nullptr, nullptr);
    ln_kernel<<<(int)MN, 128, 0, stream>>>(resid, lp.aln_g, lp.aln_b, features);
  }
}
